// BasedLinearAttention_30743375905060
// MI455X (gfx1250) — matrix operand order repaired and checked by reading, not compiled, not run on hardware
//
#include <hip/hip_runtime.h>
#include <hip/hip_bf16.h>

// ---------------------------------------------------------------------------
// Types for CDNA5 WMMA (gfx1250, wave32)
// ---------------------------------------------------------------------------
typedef __bf16 bf16_t;
typedef __attribute__((ext_vector_type(16))) __bf16 v16bf;
typedef __attribute__((ext_vector_type(8)))  float  v8f;
typedef __attribute__((ext_vector_type(4)))  int    v4i_t;

#define WMMA_BF16(a, b, c)                                                   \
  __builtin_amdgcn_wmma_f32_16x16x32_bf16(false, (a), false, (b), (short)0,  \
                                          (c), false, false)

// Problem constants (match the reference)
#define B_SZ   2
#define L_SEQ  2048
#define H_DIM  1536
#define NHEAD  12
#define FDIM   16
#define HDDIM  128
#define DPHI   288   // 273 padded up to a multiple of 32 (WMMA K step)
#define DPHI_R 273
#define CHK    64
#define MTOK   (B_SZ * L_SEQ)   // 4096 token rows

// Optional CDNA5 async global->LDS staging (ASYNCcnt-tracked DMA)
#if defined(__has_builtin)
#  if __has_builtin(__builtin_amdgcn_global_load_async_to_lds_b128) && \
      __has_builtin(__builtin_amdgcn_s_wait_asynccnt)
#    define HAVE_ASYNC_LDS 1
#  endif
#endif
#ifndef HAVE_ASYNC_LDS
#  define HAVE_ASYNC_LDS 0
#endif
#if HAVE_ASYNC_LDS
// builtin signature (from compiler diagnostic): params are `int4*` pointers
#  define TO_GLOBAL(p) ((v4i_t*)(p))
#  define TO_LDS(p)    ((v4i_t*)(p))
#endif

// ---------------------------------------------------------------------------
// Fragment loaders (wave32); A and B share one contiguous k order, k = 16*(lane>>4) + elem.
//   A (16x32 bf16): lane<16 rows M=lane, elems 0..7 = K 0..7, 8..15 = K 8..15
//                   lane>=16 rows M=lane-16, elems 0..7 = K 16..23, 8..15=K 24..31
//   B (32x16 bf16): lane holds one column N=lane%16; K ascending,
//                   K base 0 for lanes 0..15, 16 for lanes 16..31.
//   C/D (16x16 f32): elem i -> M = i + 8*(lane>>4), N = lane&15.
// "B" operands are stored column-contiguous ("Bt[n][k]") so each lane reads a
// contiguous run. f32 variants convert at load (v_cvt_pk_bf16_f32 co-executes
// with the XDL WMMA pipe).
// ---------------------------------------------------------------------------
static __device__ inline v16bf frag_a_bf(const bf16_t* base, int ld, int r0,
                                         int k0, int lane) {
  const int half = lane >> 4;
  const bf16_t* p = base + (size_t)(r0 + (lane & 15)) * ld + k0;
  v16bf f;
#pragma unroll
  for (int i = 0; i < 8; ++i) f[i] = p[half * 16 + i];
#pragma unroll
  for (int i = 0; i < 8; ++i) f[8 + i] = p[half * 16 + 8 + i];
  return f;
}

static __device__ inline v16bf frag_a_f32(const float* base, int ld, int r0,
                                          int k0, int lane) {
  const int half = lane >> 4;
  const float* p = base + (size_t)(r0 + (lane & 15)) * ld + k0;
  v16bf f;
#pragma unroll
  for (int i = 0; i < 8; ++i) f[i] = (bf16_t)p[half * 16 + i];
#pragma unroll
  for (int i = 0; i < 8; ++i) f[8 + i] = (bf16_t)p[half * 16 + 8 + i];
  return f;
}

static __device__ inline v16bf frag_bt_bf(const bf16_t* base, int ld, int n0,
                                          int k0, int lane) {
  const bf16_t* p =
      base + (size_t)(n0 + (lane & 15)) * ld + k0 + (lane >> 4) * 16;
  v16bf f;
#pragma unroll
  for (int i = 0; i < 16; ++i) f[i] = p[i];
  return f;
}

static __device__ inline v16bf frag_bt_f32(const float* base, int ld, int n0,
                                           int k0, int lane) {
  const float* p =
      base + (size_t)(n0 + (lane & 15)) * ld + k0 + (lane >> 4) * 16;
  v16bf f;
#pragma unroll
  for (int i = 0; i < 16; ++i) f[i] = (bf16_t)p[i];
  return f;
}

// ---------------------------------------------------------------------------
// GEMM helpers: 64x64 f32 tile staged per matrix; 256 threads move 16 floats
// each (4x float4) -> pure global_load_b128 / ds_store_b128 traffic.
// ---------------------------------------------------------------------------
static __device__ inline void gemm_load_regs(float4* ra, float4* rw,
                                             const float* A, const float* W,
                                             int m0, int n0, int K, int k0,
                                             int tid) {
#pragma unroll
  for (int it = 0; it < 4; ++it) {
    int idx4 = tid + it * 256;
    int r = idx4 >> 4;
    int c4 = (idx4 & 15) * 4;
    ra[it] = *(const float4*)&A[(size_t)(m0 + r) * K + k0 + c4];
    rw[it] = *(const float4*)&W[(size_t)(n0 + r) * K + k0 + c4];
  }
}

static __device__ inline void gemm_store_regs(const float4* ra,
                                              const float4* rw, float* As,
                                              float* Ws, int tid) {
#pragma unroll
  for (int it = 0; it < 4; ++it) {
    int idx4 = tid + it * 256;
    int r = idx4 >> 4;
    int c4 = (idx4 & 15) * 4;
    *(float4*)&As[r * 64 + c4] = ra[it];
    *(float4*)&Ws[r * 64 + c4] = rw[it];
  }
}

static __device__ inline void gemm_compute_step(const float* As,
                                                const float* Ws, int mi0,
                                                int ni0, int mi1, int ni1,
                                                int lane, v8f& acc0,
                                                v8f& acc1) {
#pragma unroll
  for (int kk = 0; kk < 64; kk += 32) {
    v16bf a0 = frag_a_f32(As, 64, mi0, kk, lane);
    v16bf b0 = frag_bt_f32(Ws, 64, ni0, kk, lane);
    acc0 = WMMA_BF16(a0, b0, acc0);
    v16bf a1 = frag_a_f32(As, 64, mi1, kk, lane);
    v16bf b1 = frag_bt_f32(Ws, 64, ni1, kk, lane);
    acc1 = WMMA_BF16(a1, b1, acc1);
  }
}

// ---------------------------------------------------------------------------
// Generic GEMM: C[M,N] = A[M,K] @ W[N,K]^T   (fp32 in/out, bf16 WMMA inner)
// Block = 256 threads (8 waves), 64x64 C tile, double-buffered 64-wide
// K staging. Requires M,N,K multiples of 64 (true for all uses here).
// ---------------------------------------------------------------------------
__global__ __launch_bounds__(256) void gemm_nt_wmma(
    const float* __restrict__ A, const float* __restrict__ W,
    float* __restrict__ C, int M, int N, int K) {
  __shared__ float As[2][64 * 64];
  __shared__ float Ws[2][64 * 64];
  const int tid = threadIdx.x;
  const int lane = tid & 31;
  const int wave = tid >> 5;
  const int m0 = blockIdx.y * 64;
  const int n0 = blockIdx.x * 64;

  v8f acc0 = {0, 0, 0, 0, 0, 0, 0, 0};
  v8f acc1 = {0, 0, 0, 0, 0, 0, 0, 0};
  const int t0 = wave * 2, t1 = wave * 2 + 1;
  const int mi0 = (t0 >> 2) * 16, ni0 = (t0 & 3) * 16;
  const int mi1 = (t1 >> 2) * 16, ni1 = (t1 & 3) * 16;

#if HAVE_ASYNC_LDS
  // CDNA5 async global->LDS pipeline: next tile's DMA (ASYNCcnt) overlaps
  // current tile's WMMAs. 4 b128 copies per matrix per thread = 8 per stage.
  {
#pragma unroll
    for (int it = 0; it < 4; ++it) {
      int idx4 = tid + it * 256;
      int r = idx4 >> 4;
      int c4 = (idx4 & 15) * 4;
      __builtin_amdgcn_global_load_async_to_lds_b128(
          TO_GLOBAL(&A[(size_t)(m0 + r) * K + 0 + c4]),
          TO_LDS(&As[0][r * 64 + c4]), 0, 0);
      __builtin_amdgcn_global_load_async_to_lds_b128(
          TO_GLOBAL(&W[(size_t)(n0 + r) * K + 0 + c4]),
          TO_LDS(&Ws[0][r * 64 + c4]), 0, 0);
    }
  }
  int cur = 0;
#pragma unroll 1
  for (int k0 = 0; k0 < K; k0 += 64) {
    const bool has_next = (k0 + 64) < K;
    if (has_next) {
#pragma unroll
      for (int it = 0; it < 4; ++it) {
        int idx4 = tid + it * 256;
        int r = idx4 >> 4;
        int c4 = (idx4 & 15) * 4;
        __builtin_amdgcn_global_load_async_to_lds_b128(
            TO_GLOBAL(&A[(size_t)(m0 + r) * K + k0 + 64 + c4]),
            TO_LDS(&As[cur ^ 1][r * 64 + c4]), 0, 0);
        __builtin_amdgcn_global_load_async_to_lds_b128(
            TO_GLOBAL(&W[(size_t)(n0 + r) * K + k0 + 64 + c4]),
            TO_LDS(&Ws[cur ^ 1][r * 64 + c4]), 0, 0);
      }
      __builtin_amdgcn_s_wait_asynccnt(8);  // current tile's copies done
    } else {
      __builtin_amdgcn_s_wait_asynccnt(0);
    }
    __syncthreads();  // all waves' copies of buf[cur] visible
    gemm_compute_step(&As[cur][0], &Ws[cur][0], mi0, ni0, mi1, ni1, lane,
                      acc0, acc1);
    __syncthreads();  // buf[cur] free before it becomes the next DMA target
    cur ^= 1;
  }
#else
  // Register-pipelined fallback: global loads for the next k-tile are in
  // flight while WMMAs consume the current LDS tile.
  float4 ra[4], rw[4];
  gemm_load_regs(ra, rw, A, W, m0, n0, K, 0, tid);
  gemm_store_regs(ra, rw, &As[0][0], &Ws[0][0], tid);
  __syncthreads();
  int cur = 0;
#pragma unroll 1
  for (int k0 = 0; k0 < K; k0 += 64) {
    const bool has_next = (k0 + 64) < K;
    if (has_next) {
      gemm_load_regs(ra, rw, A, W, m0, n0, K, k0 + 64, tid);
      if (k0 + 128 < K) {  // warm L2 for the tile after next
        __builtin_prefetch(&A[(size_t)(m0 + (tid >> 2)) * K + k0 + 128 + (tid & 3) * 16], 0, 0);
        __builtin_prefetch(&W[(size_t)(n0 + (tid >> 2)) * K + k0 + 128 + (tid & 3) * 16], 0, 0);
      }
    }
    gemm_compute_step(&As[cur][0], &Ws[cur][0], mi0, ni0, mi1, ni1, lane,
                      acc0, acc1);
    if (has_next)
      gemm_store_regs(ra, rw, &As[cur ^ 1][0], &Ws[cur ^ 1][0], tid);
    __syncthreads();
    cur ^= 1;
  }
#endif

  const int half = lane >> 4, nn = lane & 15;
#pragma unroll
  for (int i = 0; i < 8; ++i) {
    int r = i + half * 8;
    C[(size_t)(m0 + mi0 + r) * N + n0 + ni0 + nn] = acc0[i];
    C[(size_t)(m0 + mi1 + r) * N + n0 + ni1 + nn] = acc1[i];
  }
}

// ---------------------------------------------------------------------------
// Chunked linear attention core. One workgroup per (b,h); the 288x128 f32
// running K^T V state lives entirely in the 320KB WGP LDS (147KB) so the
// 32-chunk serial scan never touches HBM. 8 waves / 256 threads.
// Dynamic LDS layout (bytes):
//   qphi  bf16[64*288]   36864
//   kphi  bf16[64*288]   36864
//   kphiT bf16[288*64]   36864
//   vT    bf16[128*64]   16384
//   sS    bf16[64*64]     8192
//   kvT   f32 [128*288] 147456   (kvT[n][d] = KVstate[d][n])
//   ks    f32 [288]       1152
//   zz    f32 [64]         256
//   zpart f32 [256]       1024
//   qraw  f32 [64*16]     4096
//   kraw  f32 [64*16]     4096
//   total               293248  (< 320KB WGP LDS)
// ---------------------------------------------------------------------------
#define ATTN_LDS_BYTES 293248

__global__ __launch_bounds__(256) void based_attn_core(
    const float* __restrict__ Q, const float* __restrict__ Kx,
    const float* __restrict__ V, float* __restrict__ O) {
  extern __shared__ char smem_raw[];
  bf16_t* qphi  = (bf16_t*)smem_raw;          // 64x288
  bf16_t* kphi  = qphi + 64 * DPHI;           // 64x288
  bf16_t* kphiT = kphi + 64 * DPHI;           // 288x64
  bf16_t* vT    = kphiT + DPHI * CHK;         // 128x64  (vT[j][t] = v[t][j])
  bf16_t* sS    = vT + HDDIM * CHK;           // 64x64 masked scores
  float*  kvT   = (float*)(sS + CHK * CHK);   // 128x288
  float*  ks    = kvT + HDDIM * DPHI;         // 288
  float*  zz    = ks + DPHI;                  // 64
  float*  zpart = zz + CHK;                   // 256
  float*  qraw  = zpart + 256;                // 64x16
  float*  kraw  = qraw + CHK * FDIM;          // 64x16

  const int tid = threadIdx.x, lane = tid & 31, wave = tid >> 5;
  const int bh = blockIdx.x, b = bh / NHEAD, h = bh % NHEAD;

  // zero the running state (exclusive prefix starts at 0)
  for (int i = tid; i < HDDIM * DPHI; i += 256) kvT[i] = 0.f;
  for (int i = tid; i < DPHI; i += 256) ks[i] = 0.f;

  const float c1 = 0.5f;                   // 1/d^(1/4), d=16
  const float c2 = 0.17677669529663689f;   // 1/(sqrt(2)*sqrt(d))

#pragma unroll 1
  for (int ch = 0; ch < L_SEQ / CHK; ++ch) {
    const int row0 = b * L_SEQ + ch * CHK;
    __syncthreads();  // state update + tile reads of previous chunk complete

    // stage raw per-head q,k (64x16): one float4 per thread, fully coalesced
    {
      int t = tid >> 2, f4 = (tid & 3) * 4;
      *(float4*)&qraw[t * FDIM + f4] =
          *(const float4*)&Q[(size_t)(row0 + t) * (NHEAD * FDIM) + h * FDIM + f4];
      *(float4*)&kraw[t * FDIM + f4] =
          *(const float4*)&Kx[(size_t)(row0 + t) * (NHEAD * FDIM) + h * FDIM + f4];
    }
    // v chunk: coalesced float4 loads, transposed bf16 stores
#pragma unroll
    for (int it = 0; it < 8; ++it) {
      int idx4 = tid + it * 256;
      int t = idx4 >> 5;
      int j4 = (idx4 & 31) * 4;
      float4 vv =
          *(const float4*)&V[(size_t)(row0 + t) * (NHEAD * HDDIM) + h * HDDIM + j4];
      vT[(j4 + 0) * CHK + t] = (bf16_t)vv.x;
      vT[(j4 + 1) * CHK + t] = (bf16_t)vv.y;
      vT[(j4 + 2) * CHK + t] = (bf16_t)vv.z;
      vT[(j4 + 3) * CHK + t] = (bf16_t)vv.w;
    }
    __syncthreads();

    // Taylor feature map -> bf16 tiles (+ transposed copy of phi(k))
    for (int idx = tid; idx < CHK * DPHI; idx += 256) {
      int t = idx / DPHI, d = idx - t * DPHI;
      float qv, kv;
      if (d == 0) {
        qv = 1.f; kv = 1.f;
      } else if (d < 1 + FDIM) {
        qv = qraw[t * FDIM + d - 1] * c1;
        kv = kraw[t * FDIM + d - 1] * c1;
      } else if (d < DPHI_R) {
        int e = d - 1 - FDIM, i = e >> 4, j = e & 15;
        qv = qraw[t * FDIM + i] * qraw[t * FDIM + j] * c2;
        kv = kraw[t * FDIM + i] * kraw[t * FDIM + j] * c2;
      } else {
        qv = 0.f; kv = 0.f;  // padding 273..287
      }
      qphi[t * DPHI + d] = (bf16_t)qv;
      bf16_t kb = (bf16_t)kv;
      kphi[t * DPHI + d] = kb;
      kphiT[d * CHK + t] = kb;
    }
    __syncthreads();

    // S = phi(q) @ phi(k)^T  (64x64, K=288), causal-masked into LDS
    {
      const int half = lane >> 4, nn = lane & 15;
#pragma unroll
      for (int tt = 0; tt < 2; ++tt) {
        int tile = wave * 2 + tt;
        int mi = (tile >> 2) * 16, ni = (tile & 3) * 16;
        v8f acc = {0, 0, 0, 0, 0, 0, 0, 0};
#pragma unroll 1
        for (int kk = 0; kk < DPHI; kk += 32) {
          v16bf a = frag_a_bf(qphi, DPHI, mi, kk, lane);
          v16bf bb = frag_bt_bf(kphi, DPHI, ni, kk, lane);
          acc = WMMA_BF16(a, bb, acc);
        }
#pragma unroll
        for (int i = 0; i < 8; ++i) {
          int cr = mi + i + half * 8, cm = ni + nn;
          sS[cr * CHK + cm] = (cm <= cr) ? (bf16_t)acc[i] : (bf16_t)0.f;
        }
      }
    }
    __syncthreads();

    // O = S@V + phi(q)@KVstate  (64x128); z partials across all 256 threads
    v8f oacc[4];
#pragma unroll
    for (int tt = 0; tt < 4; ++tt) {
      v8f zer = {0, 0, 0, 0, 0, 0, 0, 0};
      oacc[tt] = zer;
    }
#pragma unroll 1
    for (int kk = 0; kk < DPHI; kk += 32) {
#pragma unroll
      for (int tt = 0; tt < 4; ++tt) {
        int tile = wave * 4 + tt;
        int mi = (tile >> 3) * 16, ni = (tile & 7) * 16;
        v16bf a = frag_a_bf(qphi, DPHI, mi, kk, lane);
        v16bf bb = frag_bt_f32(kvT, DPHI, ni, kk, lane);
        oacc[tt] = WMMA_BF16(a, bb, oacc[tt]);
      }
    }
#pragma unroll
    for (int kk = 0; kk < CHK; kk += 32) {
#pragma unroll
      for (int tt = 0; tt < 4; ++tt) {
        int tile = wave * 4 + tt;
        int mi = (tile >> 3) * 16, ni = (tile & 7) * 16;
        v16bf a = frag_a_bf(sS, CHK, mi, kk, lane);
        v16bf bb = frag_bt_bf(vT, CHK, ni, kk, lane);
        oacc[tt] = WMMA_BF16(a, bb, oacc[tt]);
      }
    }
    {  // z = rowsum(masked S) + phi(q).ks ; 4-way split per row
      int c = tid & 63, p = tid >> 6;
      float acc = 0.f;
      for (int m = p; m <= c; m += 4) acc += (float)sS[c * CHK + m];
      for (int d = p; d < DPHI_R; d += 4)
        acc += (float)qphi[c * DPHI + d] * ks[d];
      zpart[tid] = acc;
    }
    __syncthreads();
    if (tid < CHK)
      zz[tid] = zpart[tid] + zpart[64 + tid] + zpart[128 + tid] +
                zpart[192 + tid];
    __syncthreads();

    // normalize + store O (layout (B,L,NH*HD) for the output projection)
    {
      const int half = lane >> 4, nn = lane & 15;
#pragma unroll
      for (int tt = 0; tt < 4; ++tt) {
        int tile = wave * 4 + tt;
        int mi = (tile >> 3) * 16, ni = (tile & 7) * 16;
#pragma unroll
        for (int i = 0; i < 8; ++i) {
          int cr = mi + i + half * 8;
          float val = oacc[tt][i] / (zz[cr] + 1e-12f);
          O[(size_t)(row0 + cr) * (NHEAD * HDDIM) + h * HDDIM + ni + nn] = val;
        }
      }
    }

    // KVstate^T[n][d] += sum_t v[t][n]*phi(k)[t][d]  (128x288, K=64)
    {
      const int half = lane >> 4, nn = lane & 15;
      const int nb = wave * 16;
#pragma unroll 1
      for (int j = 0; j < DPHI / 16; ++j) {
        int db = j * 16;
        v8f cc;
#pragma unroll
        for (int i = 0; i < 8; ++i)
          cc[i] = kvT[(size_t)(nb + i + half * 8) * DPHI + db + nn];
#pragma unroll
        for (int kk = 0; kk < CHK; kk += 32) {
          v16bf a = frag_a_bf(vT, CHK, nb, kk, lane);
          v16bf bb = frag_bt_bf(kphiT, CHK, db, kk, lane);
          cc = WMMA_BF16(a, bb, cc);
        }
#pragma unroll
        for (int i = 0; i < 8; ++i)
          kvT[(size_t)(nb + i + half * 8) * DPHI + db + nn] = cc[i];
      }
    }
    // ks[d] += sum_t phi(k)[t][d]
    for (int d = tid; d < DPHI_R; d += 256) {
      float ssum = 0.f;
      for (int t = 0; t < CHK; ++t) ssum += (float)kphi[t * DPHI + d];
      ks[d] += ssum;
    }
  }
}

// ---------------------------------------------------------------------------
// Host-side launcher
// ---------------------------------------------------------------------------
extern "C" void kernel_launch(void* const* d_in, const int* in_sizes, int n_in,
                              void* d_out, int out_size, void* d_ws,
                              size_t ws_size, hipStream_t stream) {
  (void)in_sizes; (void)n_in; (void)out_size; (void)ws_size;
  const float* hs = (const float*)d_in[0];  // (B,L,H) = (4096,1536)
  const float* Wq = (const float*)d_in[1];  // (192,1536)
  const float* Wk = (const float*)d_in[2];  // (192,1536)
  const float* Wv = (const float*)d_in[3];  // (1536,1536)
  const float* Wo = (const float*)d_in[4];  // (1536,1536)
  float* out = (float*)d_out;               // (4096,1536)

  // workspace carve (floats): q 3MB, k 3MB, v 24MB, o 24MB
  float* qb = (float*)d_ws;
  float* kb = qb + (size_t)MTOK * (NHEAD * FDIM);
  float* vb = kb + (size_t)MTOK * (NHEAD * FDIM);
  float* ob = vb + (size_t)MTOK * (NHEAD * HDDIM);

  dim3 blk(256);
  // q/k/v projections
  gemm_nt_wmma<<<dim3((NHEAD * FDIM) / 64, MTOK / 64), blk, 0, stream>>>(
      hs, Wq, qb, MTOK, NHEAD * FDIM, H_DIM);
  gemm_nt_wmma<<<dim3((NHEAD * FDIM) / 64, MTOK / 64), blk, 0, stream>>>(
      hs, Wk, kb, MTOK, NHEAD * FDIM, H_DIM);
  gemm_nt_wmma<<<dim3((NHEAD * HDDIM) / 64, MTOK / 64), blk, 0, stream>>>(
      hs, Wv, vb, MTOK, NHEAD * HDDIM, H_DIM);
  // attention core: one WG per (b,h), state resident in LDS
  based_attn_core<<<dim3(B_SZ * NHEAD), blk, ATTN_LDS_BYTES, stream>>>(
      qb, kb, vb, ob);
  // output projection
  gemm_nt_wmma<<<dim3(H_DIM / 64, MTOK / 64), blk, 0, stream>>>(
      ob, Wo, out, MTOK, H_DIM, H_DIM);
}
